// Model_39676907883613
// MI455X (gfx1250) — matrix operand order repaired and checked by reading, not compiled, not run on hardware
//
#include <hip/hip_runtime.h>

// CDNA5 (gfx1250) fused attention, FlashAttention-2 style, bf16 WMMA + fp32 accumulate.
// B=2, H=16, S=2048, D=128. Double-buffered K/V tiles; K via async global->LDS DMA
// (ASYNCcnt), DPP row_ror butterfly reductions, one-ahead fragment pipelining.

typedef __attribute__((ext_vector_type(16))) __bf16 v16bf;
typedef __attribute__((ext_vector_type(8)))  __bf16 v8bf;
typedef __attribute__((ext_vector_type(2)))  __bf16 v2bf;
typedef __attribute__((ext_vector_type(8)))  float  v8f;
typedef __attribute__((ext_vector_type(4)))  int    v4i;

#define S_LEN 2048
#define DH    128
#define BR    64          // query rows per block (4 waves x 16)
#define BC    32          // keys per inner tile
#define NKB   (S_LEN / BC)

#if defined(__has_builtin)
#if __has_builtin(__builtin_amdgcn_global_load_async_to_lds_b128)
#define USE_ASYNC_LDS 1
#endif
#endif
#ifndef USE_ASYNC_LDS
#define USE_ASYNC_LDS 0
#endif

#define AS1 __attribute__((address_space(1)))
#define AS3 __attribute__((address_space(3)))

// 16-lane DPP rotate-reduce (rows of 16 == our reduction groups on wave32)
template <int CTRL>
__device__ __forceinline__ float ror_lane(float x) {
    int xi = __builtin_bit_cast(int, x);
    int r  = __builtin_amdgcn_update_dpp(xi, xi, CTRL, 0xf, 0xf, true);
    return __builtin_bit_cast(float, r);
}
__device__ __forceinline__ float rowmax16(float x) {
    x = fmaxf(x, ror_lane<0x128>(x));  // row_ror:8
    x = fmaxf(x, ror_lane<0x124>(x));  // row_ror:4
    x = fmaxf(x, ror_lane<0x122>(x));  // row_ror:2
    x = fmaxf(x, ror_lane<0x121>(x));  // row_ror:1
    return x;
}
__device__ __forceinline__ float rowsum16(float x) {
    x += ror_lane<0x128>(x);
    x += ror_lane<0x124>(x);
    x += ror_lane<0x122>(x);
    x += ror_lane<0x121>(x);
    return x;
}

// fp32 LDS -> bf16 B/A fragment (16 contiguous floats, 32B aligned)
__device__ __forceinline__ v16bf ld_frag_f32(const float* p) {
    v8f f0 = *(const v8f*)(p);
    v8f f1 = *(const v8f*)(p + 8);
    v16bf r;
#pragma unroll
    for (int i = 0; i < 8; ++i) { r[i] = (__bf16)f0[i]; r[8 + i] = (__bf16)f1[i]; }
    return r;
}

__launch_bounds__(128, 1)
__global__ void fa2_wmma_bf16_kernel(const float* __restrict__ Qg,
                                     const float* __restrict__ Kg,
                                     const float* __restrict__ scp,   // scale (1 elem)
                                     const float* __restrict__ dpp,   // dropout p (1 elem)
                                     const float* __restrict__ Vg,
                                     float* __restrict__ Og)
{
    __shared__ __align__(32) float  Ksf[2][BC][DH];     // K tiles fp32 (async DMA target) 32 KB
    __shared__ __align__(32) __bf16 Vt[2][DH][BC];      // V tiles bf16, transposed       16 KB
    __shared__ __align__(32) __bf16 Ps[4][16][BC];      // per-wave P tile                 4 KB

    const int lane = threadIdx.x & 31;
    const int wv   = threadIdx.x >> 5;
    const int n    = lane & 15;            // fragment N (B) / M (A) index
    const int hi   = lane >> 4;
    const int kbb  = hi ? 16 : 0;          // B-matrix K-row base for this lane half
    const int kqa  = hi ? 16 : 0;          // A-matrix K base
    const int kpa  = hi ? 16 : 0;

    const int    nqb   = S_LEN / BR;
    const int    bh    = blockIdx.x / nqb;
    const int    qb    = blockIdx.x % nqb;
    const size_t base  = (size_t)bh * S_LEN * DH;
    const int    qrow0 = qb * BR + wv * 16;

    const float scale2 = scp[0] * 1.44269504088896340736f;  // exp2 domain
    const float dropP  = dpp[0];

    // ---- Q A-fragments: 4 x (16x32) bf16 over D=128, direct from global ----
    v16bf qf[4];
    {
        const float* qp = Qg + base + (size_t)(qrow0 + n) * DH;
#pragma unroll
        for (int dc = 0; dc < 4; ++dc) {
            const int d0 = dc * 32 + kqa;
            float4 a0 = *(const float4*)(qp + d0);
            float4 a1 = *(const float4*)(qp + d0 + 4);
            float4 b0 = *(const float4*)(qp + d0 + 8);
            float4 b1 = *(const float4*)(qp + d0 + 12);
            v16bf f;
            f[0]  = (__bf16)a0.x; f[1]  = (__bf16)a0.y; f[2]  = (__bf16)a0.z; f[3]  = (__bf16)a0.w;
            f[4]  = (__bf16)a1.x; f[5]  = (__bf16)a1.y; f[6]  = (__bf16)a1.z; f[7]  = (__bf16)a1.w;
            f[8]  = (__bf16)b0.x; f[9]  = (__bf16)b0.y; f[10] = (__bf16)b0.z; f[11] = (__bf16)b0.w;
            f[12] = (__bf16)b1.x; f[13] = (__bf16)b1.y; f[14] = (__bf16)b1.z; f[15] = (__bf16)b1.w;
            qf[dc] = f;
        }
    }

    // tile fill: K via async DMA (fp32), V via manual transpose+convert (bf16, packed stores)
    auto fill_tiles = [&](int buf, int kb) {
        const int key = threadIdx.x >> 2;          // 0..31
        const int dch = (threadIdx.x & 3) * 32;    // 0,32,64,96
        const float* kp = Kg + base + (size_t)(kb * BC + key) * DH + dch;
        float* ld = &Ksf[buf][key][dch];
#if USE_ASYNC_LDS
#pragma unroll
        for (int i = 0; i < 8; ++i)
            __builtin_amdgcn_global_load_async_to_lds_b128(
                (AS1 v4i*)(kp + i * 4), (AS3 v4i*)(ld + i * 4), 0, 0);
#else
#pragma unroll
        for (int i = 0; i < 8; ++i)
            *(float4*)(ld + i * 4) = *(const float4*)(kp + i * 4);
#endif
        const float* vp = Vg + base + (size_t)(kb * BC) * DH + threadIdx.x;
#pragma unroll
        for (int k = 0; k < BC; k += 2) {
            float a = vp[(size_t)k * DH];
            float b = vp[(size_t)(k + 1) * DH];
            v2bf pk; pk[0] = (__bf16)a; pk[1] = (__bf16)b;
            *(v2bf*)&Vt[buf][threadIdx.x][k] = pk;
        }
    };

    const v8f zero8 = {0.f, 0.f, 0.f, 0.f, 0.f, 0.f, 0.f, 0.f};
    v8f o[8];
#pragma unroll
    for (int t = 0; t < 8; ++t) o[t] = zero8;
    float m_state[8], l_state[8];
#pragma unroll
    for (int r = 0; r < 8; ++r) { m_state[r] = -3.0e38f; l_state[r] = 0.f; }

    // prologue: fill buffer 0
    fill_tiles(0, 0);
    asm volatile("s_wait_asynccnt 0" ::: "memory");
    __syncthreads();

    for (int kb = 0; kb < NKB; ++kb) {
        const int cur = kb & 1;
        if (kb + 1 < NKB) fill_tiles(cur ^ 1, kb + 1);   // overlap DMA with compute

        // ---- scores: S = Q.K^T, two 16-key subtiles, one-ahead fragment pipeline ----
        const float* ksb = &Ksf[cur][0][0];
        v8f c0 = zero8, c1 = zero8;
        v16bf kc = ld_frag_f32(ksb + (size_t)n * DH + kbb);   // i=0: sub 0, dc 0
#pragma unroll
        for (int i = 0; i < 8; ++i) {                          // i = dc*2 + sub
            v16bf kn = kc;
            if (i < 7) {
                const int s2 = (i + 1) & 1, d2 = (i + 1) >> 1;
                kn = ld_frag_f32(ksb + (size_t)(s2 * 16 + n) * DH + d2 * 32 + kbb);
            }
            const int dc = i >> 1;
            if ((i & 1) == 0)
                c0 = __builtin_amdgcn_wmma_f32_16x16x32_bf16(false, qf[dc], false, kc, (short)0, c0, false, false);
            else
                c1 = __builtin_amdgcn_wmma_f32_16x16x32_bf16(false, qf[dc], false, kc, (short)0, c1, false, false);
            kc = kn;
        }

        // ---- online softmax (exp2 domain); DPP row reductions, no LDS traffic ----
#pragma unroll
        for (int r = 0; r < 8; ++r) {
            float s0 = c0[r] * scale2;
            float s1 = c1[r] * scale2;
            float rm = rowmax16(fmaxf(s0, s1));
            float mn = fmaxf(m_state[r], rm);
            float p0 = exp2f(s0 - mn);
            float p1 = exp2f(s1 - mn);
            float al = exp2f(m_state[r] - mn);
            float rs = rowsum16(p0 + p1);
            l_state[r] = l_state[r] * al + rs;
            m_state[r] = mn;
            Ps[wv][r + 8 * hi][n]      = (__bf16)p0;
            Ps[wv][r + 8 * hi][16 + n] = (__bf16)p1;
#pragma unroll
            for (int t = 0; t < 8; ++t) o[t][r] *= al;
        }

        // wave-private LDS RAW fence (lockstep wave; no block barrier needed)
        asm volatile("s_wait_dscnt 0" ::: "memory");

        // ---- P A-fragment (16x32 bf16) ----
        v16bf pf;
        {
            v8bf plo = *(const v8bf*)&Ps[wv][n][kpa];
            v8bf phi = *(const v8bf*)&Ps[wv][n][kpa + 8];
#pragma unroll
            for (int i = 0; i < 8; ++i) { pf[i] = plo[i]; pf[8 + i] = phi[i]; }
        }

        // ---- O += P.V : 8 WMMAs over D, one-ahead V-fragment pipeline ----
        v16bf vc = *(const v16bf*)&Vt[cur][n][kbb];            // t=0
#pragma unroll
        for (int t = 0; t < 8; ++t) {
            v16bf vn = vc;
            if (t < 7) vn = *(const v16bf*)&Vt[cur][(t + 1) * 16 + n][kbb];
            o[t] = __builtin_amdgcn_wmma_f32_16x16x32_bf16(false, pf, false, vc, (short)0, o[t], false, false);
            vc = vn;
        }

        // next tile's async K must be resident before anyone reads it
        asm volatile("s_wait_asynccnt 0" ::: "memory");
        __syncthreads();
    }

    // ---- epilogue: normalize by l and (1-p), store fp32 ----
    const float invDrop = 1.0f / (1.0f - dropP);
#pragma unroll
    for (int r = 0; r < 8; ++r) {
        float inv = invDrop / l_state[r];
        float* op = Og + base + (size_t)(qrow0 + r + 8 * hi) * DH;
#pragma unroll
        for (int t = 0; t < 8; ++t)
            op[t * 16 + n] = o[t][r] * inv;
    }
}

extern "C" void kernel_launch(void* const* d_in, const int* in_sizes, int n_in,
                              void* d_out, int out_size, void* d_ws, size_t ws_size,
                              hipStream_t stream) {
    (void)in_sizes; (void)n_in; (void)out_size; (void)d_ws; (void)ws_size;
    const float* Q  = (const float*)d_in[0];
    const float* K  = (const float*)d_in[1];
    const float* sc = (const float*)d_in[2];
    const float* dp = (const float*)d_in[3];
    const float* V  = (const float*)d_in[4];
    float* O = (float*)d_out;

    const int B = 2, H = 16;
    dim3 grid(B * H * (S_LEN / BR));   // 1024 blocks
    dim3 block(128);                   // 4 waves (wave32)
    fa2_wmma_bf16_kernel<<<grid, block, 0, stream>>>(Q, K, sc, dp, V, O);
}
